// Renet2d_59837484368062
// MI455X (gfx1250) — matrix operand order repaired and checked by reading, compiled, not run on hardware
//
#include <hip/hip_runtime.h>

typedef __attribute__((ext_vector_type(16))) __bf16 v16bf;
typedef __attribute__((ext_vector_type(8)))  float  v8f;

#define HIDN 256
#define TLEN 128

__device__ __forceinline__ float sigmoidf_(float x) {
    return 1.0f / (1.0f + __expf(-x));
}

// ---------------------------------------------------------------------------
// Pack x (B=8,C=64,H=128,W=128) f32 -> bf16 WMMA A-tile layout for the
// vertical sweep: [seqTile(64)][t(128)][kc(2)][lane(32)][slot(16)]
// A-layout (16-bit A 16x32): lane = m + 16*khalf ; slot s -> k = block*16 +
// khalf*8 + (s&7), block = s>>3.
// ---------------------------------------------------------------------------
__global__ void __launch_bounds__(256)
k_pack_x(const float* __restrict__ x, __bf16* __restrict__ dst) {
    int idx  = blockIdx.x * 256 + threadIdx.x;          // 8,388,608 elements
    int slot = idx & 15;
    int lane = (idx >> 4) & 31;
    int kc   = (idx >> 9) & 1;
    int t    = (idx >> 10) & 127;
    int seqTile = idx >> 17;                            // 0..63
    int m     = lane & 15;
    int khalf = lane >> 4;
    int block = slot >> 3;
    int k = kc * 32 + block * 16 + khalf * 8 + (slot & 7);   // channel 0..63
    int seq = seqTile * 16 + m;                              // (b,h)
    int b = seq >> 7, hh = seq & 127;
    dst[idx] = (__bf16)x[(((long)(b * 64 + k) * 128 + hh) * 128) + t];
}

// ---------------------------------------------------------------------------
// Pack W = [Wih | Whh] (G x K) f32 -> bf16 WMMA B-tile layout:
// [nt][kc][lane(32)][slot(16)], B 32x16: lane = n + 16*khi, k = (slot>>3)*16+khi*8+(slot&7).
// ---------------------------------------------------------------------------
__global__ void __launch_bounds__(256)
k_pack_w(const float* __restrict__ Wih, const float* __restrict__ Whh,
         int D, int KC, __bf16* __restrict__ dst, int total) {
    int idx = blockIdx.x * 256 + threadIdx.x;
    if (idx >= total) return;
    int slot = idx & 15;
    int lane = (idx >> 4) & 31;
    int tile = idx >> 9;
    int kc   = tile % KC;
    int nt   = tile / KC;
    int n = nt * 16 + (lane & 15);
    int k = kc * 32 + (slot >> 3) * 16 + (lane >> 4) * 8 + (slot & 7);
    float v = (k < D) ? Wih[(long)n * D + k] : Whh[(long)n * HIDN + (k - D)];
    dst[idx] = (__bf16)v;
}

// ---------------------------------------------------------------------------
// Fused bidirectional LSTM sweep.  One workgroup = 16 sequences, 8 waves.
// Wave w owns hid slice [32w,32w+32): its i/f/g/o gate tiles (8 WMMA n-tiles)
// accumulate in registers; c and the nonlinearity never leave VGPRs.  h is
// rebroadcast each step as bf16 A-tiles through 8KB of LDS.
// K = [x_t (KCX chunks, global, pre-packed) | h (8 chunks, LDS)] fuses the
// input projection into the recurrent GEMM.  KCX/MODE are template constants
// so the K loop fully unrolls: all loads become base+immediate, and each
// chunk issues its 8 B-tile loads as one clause ahead of the 8 WMMAs.
// ---------------------------------------------------------------------------
template <int KCX, int MODE>
__global__ void __launch_bounds__(256)
k_lstm(const __bf16* __restrict__ xpack,
       const __bf16* __restrict__ wfw, const __bf16* __restrict__ wbw,
       const float* __restrict__ bfw, const float* __restrict__ bbw,
       __bf16* __restrict__ outpack) {
    constexpr int KC = KCX + 8;
    __shared__ v16bf hAv[8][32];                 // 8 K-chunks of h as A-tiles

    const int seqTile = blockIdx.x;
    const int dir     = blockIdx.y;
    const int tid     = threadIdx.x;
    const int wave    = tid >> 5;
    const int lane    = tid & 31;
    const __bf16* __restrict__ wpack = dir ? wbw : wfw;
    const float*  __restrict__ bias  = dir ? bbw : bfw;

    {   // h starts at zero
        __bf16* hz = (__bf16*)hAv;
        for (int i = tid; i < 8 * 32 * 16; i += 256) hz[i] = (__bf16)0.0f;
    }
    __syncthreads();

    const int ncol = lane & 15;         // n within a tile / D-layout column
    const int mhi  = lane >> 4;         // contributes +8 to D-layout row

    // per-gate-tile weight base pointers and biases (g: I0 I1 F0 F1 G0 G1 O0 O1)
    const __bf16* wbase[8];
    float bgate[8];
#pragma unroll
    for (int g = 0; g < 8; ++g) {
        int nt = (g >> 1) * 16 + 2 * wave + (g & 1);
        wbase[g] = wpack + ((long)nt * KC * 32 + lane) * 16;
        bgate[g] = bias[nt * 16 + ncol];
    }

    v8f c[2] = {{}, {}};

    for (int t = 0; t < TLEN; ++t) {
        const int tt = dir ? (TLEN - 1 - t) : t;   // position in the sequence

        v8f acc[8];
#pragma unroll
        for (int g = 0; g < 8; ++g)
#pragma unroll
            for (int r = 0; r < 8; ++r) acc[g][r] = bgate[g];

        const __bf16* xbase =
            xpack + (((long)seqTile * TLEN + tt) * KCX) * 512 + (long)lane * 16;
        if (t + 1 < TLEN) {             // prefetch next step's x A-tiles
            const int tn = dir ? (TLEN - 2 - t) : (t + 1);
            __builtin_prefetch(
                xpack + (((long)seqTile * TLEN + tn) * KCX) * 512 + (long)lane * 16,
                0, 1);
        }

#pragma unroll
        for (int kc = 0; kc < KC; ++kc) {
            v16bf a;
            if (kc < KCX) a = *(const v16bf*)(xbase + kc * 512);
            else          a = hAv[kc - KCX][lane];
            v16bf bt[8];
#pragma unroll
            for (int g = 0; g < 8; ++g)
                bt[g] = *(const v16bf*)(wbase[g] + kc * 512);
#pragma unroll
            for (int g = 0; g < 8; ++g)
                acc[g] = __builtin_amdgcn_wmma_f32_16x16x32_bf16(
                    false, a, false, bt[g], (short)0, acc[g], false, false);
        }
        __syncthreads();     // everyone done reading hAv for this step

        // time-dependent part of the scatter address (shared by all 16 values)
        const long toff = (MODE == 0)
            ? (long)(tt >> 4) * (128L * 16 * 512) + (long)(tt & 15) * 16
            : (long)tt * (8L * 16 * 512);

        __bf16* hAe = (__bf16*)hAv;
#pragma unroll
        for (int q = 0; q < 2; ++q) {
#pragma unroll
            for (int r = 0; r < 8; ++r) {
                float iv = sigmoidf_(acc[0 + q][r]);
                float fv = sigmoidf_(acc[2 + q][r]);
                float gv = tanhf(acc[4 + q][r]);
                float ov = sigmoidf_(acc[6 + q][r]);
                float cv = fv * c[q][r] + iv * gv;
                c[q][r] = cv;
                float hv = ov * tanhf(cv);

                int m = r + 8 * mhi;                       // sequence row 0..15
                int j = 32 * wave + 16 * q + ncol;         // hid index 0..255

                // rebroadcast h as next step's A-tile (chunk = wave)
                int s     = q * 8 + (lane & 7);
                int laneA = m + 16 * ((ncol >> 3) & 1);
                hAe[(wave * 32 + laneA) * 16 + s] = (__bf16)hv;

                // t-invariant base of the consumer's packed A-tile address
                int seq   = seqTile * 16 + m;
                int kfeat = dir * HIDN + j;                // 0..511
                int ch2 = kfeat >> 5, kk2 = kfeat & 31;
                int s2  = (kk2 >> 4) * 8 + (kk2 & 7);
                int kh2 = (kk2 >> 3) & 1;
                int b   = seq >> 7;
                long cbase;
                if (MODE == 0) {       // vertical -> horizontal input
                    int hh = seq & 127;
                    cbase = (((long)b * 1024 + hh) * 16 + ch2) * 512 +
                            (long)(16 * kh2) * 16 + s2;
                } else {               // horizontal -> conv input
                    int wcol = seq & 127;
                    cbase = (((long)b * 1024 + (wcol >> 4)) * 16 + ch2) * 512 +
                            (long)((wcol & 15) + 16 * kh2) * 16 + s2;
                }
                outpack[cbase + toff] = (__bf16)hv;
            }
        }
        __syncthreads();     // hAv fully rewritten before next step reads
    }
}

// ---------------------------------------------------------------------------
// 1x1 conv as WMMA GEMM: M = 131072 spatial positions, K = 512, N = 64.
// One block = one M-tile, 4 waves = 4 N-tiles.
// ---------------------------------------------------------------------------
__global__ void __launch_bounds__(128)
k_conv(const __bf16* __restrict__ cpack, const __bf16* __restrict__ wpack,
       const float* __restrict__ bconv, float* __restrict__ out) {
    const int rowTile = blockIdx.x;                 // 0..8191
    const int wave = threadIdx.x >> 5, lane = threadIdx.x & 31;
    const int ncol = lane & 15;

    v8f acc;
    {
        float bv = bconv[wave * 16 + ncol];
#pragma unroll
        for (int r = 0; r < 8; ++r) acc[r] = bv;
    }
    const __bf16* abase = cpack + ((long)rowTile * 16 * 32 + lane) * 16;
    const __bf16* bbase = wpack + ((long)wave * 16 * 32 + lane) * 16;
#pragma unroll
    for (int kc = 0; kc < 16; kc += 4) {   // group loads 4 chunks at a time
        v16bf a0 = *(const v16bf*)(abase + (kc + 0) * 512);
        v16bf a1 = *(const v16bf*)(abase + (kc + 1) * 512);
        v16bf a2 = *(const v16bf*)(abase + (kc + 2) * 512);
        v16bf a3 = *(const v16bf*)(abase + (kc + 3) * 512);
        v16bf b0 = *(const v16bf*)(bbase + (kc + 0) * 512);
        v16bf b1 = *(const v16bf*)(bbase + (kc + 1) * 512);
        v16bf b2 = *(const v16bf*)(bbase + (kc + 2) * 512);
        v16bf b3 = *(const v16bf*)(bbase + (kc + 3) * 512);
        acc = __builtin_amdgcn_wmma_f32_16x16x32_bf16(false, a0, false, b0, (short)0, acc, false, false);
        acc = __builtin_amdgcn_wmma_f32_16x16x32_bf16(false, a1, false, b1, (short)0, acc, false, false);
        acc = __builtin_amdgcn_wmma_f32_16x16x32_bf16(false, a2, false, b2, (short)0, acc, false, false);
        acc = __builtin_amdgcn_wmma_f32_16x16x32_bf16(false, a3, false, b3, (short)0, acc, false, false);
    }
#pragma unroll
    for (int r = 0; r < 8; ++r) {
        int m = r + 8 * (lane >> 4);
        int row = rowTile * 16 + m;
        int b = row >> 14, hh = (row >> 7) & 127, w = row & 127;
        int o = wave * 16 + ncol;
        out[(((long)(b * 64 + o) * 128 + hh) * 128) + w] = acc[r];
    }
}

// ---------------------------------------------------------------------------
extern "C" void kernel_launch(void* const* d_in, const int* in_sizes, int n_in,
                              void* d_out, int out_size, void* d_ws, size_t ws_size,
                              hipStream_t stream) {
    (void)in_sizes; (void)n_in; (void)out_size; (void)ws_size;
    const float* x      = (const float*)d_in[0];
    const float* Wih_vf = (const float*)d_in[1];
    const float* Whh_vf = (const float*)d_in[2];
    const float* b_vf   = (const float*)d_in[3];
    const float* Wih_vb = (const float*)d_in[4];
    const float* Whh_vb = (const float*)d_in[5];
    const float* b_vb   = (const float*)d_in[6];
    const float* Wih_hf = (const float*)d_in[7];
    const float* Whh_hf = (const float*)d_in[8];
    const float* b_hf   = (const float*)d_in[9];
    const float* Wih_hb = (const float*)d_in[10];
    const float* Whh_hb = (const float*)d_in[11];
    const float* b_hb   = (const float*)d_in[12];
    const float* Wconv  = (const float*)d_in[13];
    const float* bconv  = (const float*)d_in[14];

    char* ws = (char*)d_ws;
    const size_t MB = 1024 * 1024;
    __bf16* xpv = (__bf16*)(ws + 0);            // 16 MB  vertical x, A-tiles
    __bf16* wvf = (__bf16*)(ws + 16 * MB);      // 640 KB vertical fwd weights
    __bf16* wvb = (__bf16*)(ws + 17 * MB);      // 640 KB vertical bwd weights
    __bf16* whf = (__bf16*)(ws + 18 * MB);      // 1.5 MB horizontal fwd
    __bf16* whb = (__bf16*)(ws + 20 * MB);      // 1.5 MB horizontal bwd
    __bf16* wcv = (__bf16*)(ws + 22 * MB);      // 64 KB  conv weights
    __bf16* hx  = (__bf16*)(ws + 24 * MB);      // 128 MB horizontal input
    __bf16* cp  = (__bf16*)(ws + 152 * MB);     // 128 MB conv input

    // pack everything to WMMA-native bf16 layouts
    k_pack_x<<<32768, 256, 0, stream>>>(x, xpv);
    k_pack_w<<<1280, 256, 0, stream>>>(Wih_vf, Whh_vf,  64, 10, wvf, 64 * 10 * 512);
    k_pack_w<<<1280, 256, 0, stream>>>(Wih_vb, Whh_vb,  64, 10, wvb, 64 * 10 * 512);
    k_pack_w<<<3072, 256, 0, stream>>>(Wih_hf, Whh_hf, 512, 24, whf, 64 * 24 * 512);
    k_pack_w<<<3072, 256, 0, stream>>>(Wih_hb, Whh_hb, 512, 24, whb, 64 * 24 * 512);
    k_pack_w<<<128,  256, 0, stream>>>(Wconv,  Wconv,  512, 16, wcv, 4 * 16 * 512);

    // vertical bidirectional sweep (both directions concurrently, blockIdx.y)
    k_lstm<2, 0><<<dim3(64, 2), 256, 0, stream>>>(xpv, wvf, wvb, b_vf, b_vb, hx);
    // horizontal bidirectional sweep
    k_lstm<16, 1><<<dim3(64, 2), 256, 0, stream>>>(hx, whf, whb, b_hf, b_hb, cp);
    // 1x1 conv
    k_conv<<<8192, 128, 0, stream>>>(cp, wcv, bconv, (float*)d_out);
}
